// GCN_7086696039015
// MI455X (gfx1250) — compile-verified, matrix operand order checked by reading, not run on hardware
//
#include <hip/hip_runtime.h>
#include <hip/hip_bf16.h>

// ---------------------------------------------------------------------------
// CDNA5 (gfx1250) GCN pipeline: bf16 WMMA GEMMs + atomic edge scatter.
// wave32 everywhere; 256-thread blocks = 8 waves.
// Activations are pre-converted to bf16 so the GEMM inner loop is pure
// b128-load + v_wmma (no in-register packing), and the bf16 A matrix
// (<=102.4 MB) stays resident in the 192 MB L2 across column-group passes.
// ---------------------------------------------------------------------------

typedef __attribute__((ext_vector_type(16))) __bf16 v16bf;
typedef __attribute__((ext_vector_type(8)))  float  v8f;

static __device__ __forceinline__ unsigned pack_bf16(float a, float b) {
  // round-to-nearest-even fp32 -> bf16, packed low|high
  unsigned ua = __float_as_uint(a);
  unsigned ub = __float_as_uint(b);
  ua += 0x7FFFu + ((ua >> 16) & 1u);
  ub += 0x7FFFu + ((ub >> 16) & 1u);
  return (ua >> 16) | (ub & 0xFFFF0000u);
}

// --------------------------- GEMM: C[N,M] = A[N,K] @ W  (+bias)(+acc)(relu) --
// A and W both bf16; W pre-transposed: Wt[m*K + k].
// One wave -> 16 rows x 64 cols (4 accumulator tiles), K-loop step 32.
__global__ __launch_bounds__(256) void gemm_wmma_bf16(
    const unsigned short* __restrict__ A, const unsigned short* __restrict__ Wt,
    const float* __restrict__ bias, float* __restrict__ C,
    int N, int K, int M, int relu, int acc)
{
  const int wave  = threadIdx.x >> 5;
  const int lane  = threadIdx.x & 31;
  const int r0    = (blockIdx.x * 8 + wave) * 16;
  if (r0 >= N) return;
  const int m0    = blockIdx.y * 64;
  const int mrow  = lane & 15;   // A row / B col within tile
  const int khalf = lane >> 4;   // which K-half this lane holds

  const v8f vzero = {0.f,0.f,0.f,0.f,0.f,0.f,0.f,0.f};
  v8f accv[4] = {vzero, vzero, vzero, vzero};

  const unsigned short* arow = A + (size_t)(r0 + mrow) * K + khalf * 8;

  for (int k0 = 0; k0 < K; k0 += 32) {
    __builtin_prefetch(arow + k0 + 32, 0, 1);           // global_prefetch_b8
    // ---- A fragment: 16x32 bf16 per ISA layout (two b128 loads) ----
    union { unsigned u[8]; v16bf v; } af;
    const uint4 a0 = *(const uint4*)(arow + k0);        // k = khalf*8 .. +7
    const uint4 a1 = *(const uint4*)(arow + k0 + 16);   // k = 16+khalf*8 .. +7
    af.u[0] = a0.x; af.u[1] = a0.y; af.u[2] = a0.z; af.u[3] = a0.w;
    af.u[4] = a1.x; af.u[5] = a1.y; af.u[6] = a1.z; af.u[7] = a1.w;
#pragma unroll
    for (int t = 0; t < 4; ++t) {
      const unsigned short* wr =
          Wt + (size_t)(m0 + t * 16 + mrow) * K + k0 + khalf * 8;
      union { unsigned u[8]; v16bf v; } bfr;
      const uint4 w0 = *(const uint4*)(wr);
      const uint4 w1 = *(const uint4*)(wr + 16);
      bfr.u[0] = w0.x; bfr.u[1] = w0.y; bfr.u[2] = w0.z; bfr.u[3] = w0.w;
      bfr.u[4] = w1.x; bfr.u[5] = w1.y; bfr.u[6] = w1.z; bfr.u[7] = w1.w;
      accv[t] = __builtin_amdgcn_wmma_f32_16x16x32_bf16(
          false, af.v, false, bfr.v, (short)0, accv[t], false, false);
    }
  }
  // ---- store C (VGPR v -> row v + 8*khalf; lane%16 -> col) ----
#pragma unroll
  for (int t = 0; t < 4; ++t) {
    const int col = m0 + t * 16 + mrow;
    const float bv = bias ? bias[col] : 0.0f;
#pragma unroll
    for (int v = 0; v < 8; ++v) {
      const int row = r0 + khalf * 8 + v;
      const size_t idx = (size_t)row * M + col;
      float val = accv[t][v] + bv;
      if (acc)  val += C[idx];
      if (relu) val = fmaxf(val, 0.0f);
      C[idx] = val;
    }
  }
}

// --------------------------- conversion / elementwise / graph kernels --------
// fp32 -> bf16, 8 elements per thread (float4 x2 in, uint4 out)
__global__ void f32_to_bf16(const float* __restrict__ src,
                            unsigned short* __restrict__ dst, size_t n8) {
  const size_t i = (size_t)blockIdx.x * 256 + threadIdx.x;
  if (i >= n8) return;
  const float4 a = ((const float4*)src)[2 * i];
  const float4 b = ((const float4*)src)[2 * i + 1];
  uint4 o;
  o.x = pack_bf16(a.x, a.y); o.y = pack_bf16(a.z, a.w);
  o.z = pack_bf16(b.x, b.y); o.w = pack_bf16(b.z, b.w);
  ((uint4*)dst)[i] = o;
}

__global__ void fill_f32(float* __restrict__ p, float v, size_t n) {
  size_t i = (size_t)blockIdx.x * 256 + threadIdx.x;
  if (i < n) p[i] = v;
}

__global__ void transpose_to_bf16(const float* __restrict__ W,
                                  unsigned short* __restrict__ Wt,
                                  int K, int M) {
  int gid = blockIdx.x * 256 + threadIdx.x;
  if (gid >= K * M) return;
  int m = gid / K, k = gid % K;
  unsigned u = __float_as_uint(W[(size_t)k * M + m]);
  u += 0x7FFFu + ((u >> 16) & 1u);
  Wt[gid] = (unsigned short)(u >> 16);
}

__global__ void deg_accum(const int* __restrict__ dst, float* __restrict__ deg, int E) {
  int e = blockIdx.x * 256 + threadIdx.x;
  if (e < E) atomicAdd(&deg[dst[e]], 1.0f);
}

__global__ void deg_to_dis(const float* __restrict__ deg, float* __restrict__ dis, int N) {
  int i = blockIdx.x * 256 + threadIdx.x;
  if (i < N) dis[i] = rsqrtf(deg[i]);   // deg >= 1 (self-loop)
}

// out[i,:] = xw[i,:]*dis[i]^2 + bias   (GCN self-loop term + bias)
__global__ __launch_bounds__(256) void gcn_selfloop(
    const float* __restrict__ xw, const float* __restrict__ dis,
    const float* __restrict__ bias, float* __restrict__ out) {
  const int i = blockIdx.x, c = threadIdx.x;
  const float d = dis[i];
  out[(size_t)i * 256 + c] = xw[(size_t)i * 256 + c] * d * d + bias[c];
}

// out[dst,:] += xw[src,:] * dis[src]*dis[dst]   (one block per edge, 256 ch)
__global__ __launch_bounds__(256) void gcn_edge_scatter(
    const float* __restrict__ xw, const float* __restrict__ dis,
    const int* __restrict__ src, const int* __restrict__ dst,
    float* __restrict__ out) {
  const int e = blockIdx.x, c = threadIdx.x;
  const int s = src[e], d = dst[e];
  const float nrm = dis[s] * dis[d];
  atomicAdd(&out[(size_t)d * 256 + c], xw[(size_t)s * 256 + c] * nrm);
}

__global__ __launch_bounds__(256) void sage_edge_scatter(
    const float* __restrict__ x, const int* __restrict__ src,
    const int* __restrict__ dst, float* __restrict__ out) {
  const int e = blockIdx.x, c = threadIdx.x;
  atomicAdd(&out[(size_t)dst[e] * 256 + c], x[(size_t)src[e] * 256 + c]);
}

// x0 = relu(h)+relu(g);  g1 = relu(g)+g
__global__ __launch_bounds__(256) void combine_streams(
    const float* __restrict__ h, const float* __restrict__ g,
    float* __restrict__ x0, float* __restrict__ g1) {
  const size_t idx = (size_t)blockIdx.x * 256 + threadIdx.x;
  const float hv = h[idx], gv = g[idx];
  const float hr = fmaxf(hv, 0.f), gr = fmaxf(gv, 0.f);
  x0[idx] = hr + gr;
  g1[idx] = gr + gv;
}

// agg[i,:] /= max(deg[i]-1, 1)   (SAGE mean; deg includes the +1 self-loop)
__global__ __launch_bounds__(256) void sage_mean(
    float* __restrict__ agg, const float* __restrict__ deg) {
  const int i = blockIdx.x, c = threadIdx.x;
  agg[(size_t)i * 256 + c] /= fmaxf(deg[i] - 1.0f, 1.0f);
}

// Per-row (128): LayerNorm then dot with Wm2; accumulate scalar sum. 1 wave/row.
__global__ __launch_bounds__(256) void mlp_ln_dot(
    const float* __restrict__ T, const float* __restrict__ lnw,
    const float* __restrict__ lnb, const float* __restrict__ Wm2,
    const float* __restrict__ bm2, float* __restrict__ sum, int N)
{
  const int wave = threadIdx.x >> 5;
  const int lane = threadIdx.x & 31;
  const int row  = blockIdx.x * 8 + wave;
  __shared__ float ls[8];
  float rowres = 0.0f;
  if (row < N) {
    const float* t = T + (size_t)row * 128;
    float v0 = t[lane], v1 = t[lane + 32], v2 = t[lane + 64], v3 = t[lane + 96];
    float mu = v0 + v1 + v2 + v3;
    for (int off = 16; off > 0; off >>= 1) mu += __shfl_xor(mu, off, 32);
    mu *= (1.0f / 128.0f);
    const float d0 = v0 - mu, d1 = v1 - mu, d2 = v2 - mu, d3 = v3 - mu;
    float var = d0 * d0 + d1 * d1 + d2 * d2 + d3 * d3;
    for (int off = 16; off > 0; off >>= 1) var += __shfl_xor(var, off, 32);
    var *= (1.0f / 128.0f);
    const float inv = rsqrtf(var + 1e-5f);
    float s = 0.f;
    s += (d0 * inv * lnw[lane]      + lnb[lane])      * Wm2[lane];
    s += (d1 * inv * lnw[lane + 32] + lnb[lane + 32]) * Wm2[lane + 32];
    s += (d2 * inv * lnw[lane + 64] + lnb[lane + 64]) * Wm2[lane + 64];
    s += (d3 * inv * lnw[lane + 96] + lnb[lane + 96]) * Wm2[lane + 96];
    for (int off = 16; off > 0; off >>= 1) s += __shfl_xor(s, off, 32);
    rowres = s + bm2[0];
  }
  if (lane == 0) ls[wave] = rowres;
  __syncthreads();
  if (threadIdx.x == 0) {
    float tot = 0.f;
    for (int w = 0; w < 8; ++w) tot += ls[w];
    atomicAdd(sum, tot);
  }
}

__global__ void finalize_mean(const float* __restrict__ sum,
                              float* __restrict__ out, float invc) {
  out[0] = sum[0] * invc;
}

// ---------------------------------------------------------------------------
extern "C" void kernel_launch(void* const* d_in, const int* in_sizes, int n_in,
                              void* d_out, int out_size, void* d_ws, size_t ws_size,
                              hipStream_t stream) {
  const float* x    = (const float*)d_in[0];
  const float* x1   = (const float*)d_in[1];
  const int*   eiA  = (const int*)d_in[2];
  const int*   eiB  = (const int*)d_in[3];
  const float* Wg1  = (const float*)d_in[4];
  const float* bg1  = (const float*)d_in[5];
  const float* Wg2  = (const float*)d_in[6];
  const float* bg2  = (const float*)d_in[7];
  const float* Wl1  = (const float*)d_in[8];
  const float* bl1  = (const float*)d_in[9];
  const float* Wr1  = (const float*)d_in[10];
  const float* Wl2  = (const float*)d_in[11];
  const float* bl2  = (const float*)d_in[12];
  const float* Wr2  = (const float*)d_in[13];
  const float* Wm1  = (const float*)d_in[14];
  const float* bm1  = (const float*)d_in[15];
  const float* lnw  = (const float*)d_in[16];
  const float* lnb  = (const float*)d_in[17];
  const float* Wm2  = (const float*)d_in[18];
  const float* bm2  = (const float*)d_in[19];
  float* out = (float*)d_out;

  const int N = in_sizes[0] / 512;      // 100000
  const int E = in_sizes[2] / 2;        // 600000
  const int F = 512, H = 256, H2 = 128;
  const int* srcA = eiA;       const int* dstA = eiA + E;
  const int* srcB = eiB;       const int* dstB = eiB + E;

  // ---- workspace carve-out ----
  char* base = (char*)d_ws;
  size_t off = 0;
  auto take = [&](size_t bytes) { char* p = base + off; off += (bytes + 255) & ~(size_t)255; return p; };
  const size_t NB = (size_t)N * H * sizeof(float);
  float* buf1 = (float*)take(NB);          // xw_a -> x0
  float* buf2 = (float*)take(NB);          // xw_b -> g1
  float* buf3 = (float*)take(NB);          // h    -> mean/T
  float* buf4 = (float*)take(NB);          // g    -> a2/b2
  unsigned short* abf = (unsigned short*)take((size_t)N * F * 2);  // bf16 activations
  float* degA = (float*)take((size_t)N * 4);
  float* degB = (float*)take((size_t)N * 4);
  float* disA = (float*)take((size_t)N * 4);
  float* disB = (float*)take((size_t)N * 4);
  float* sum  = (float*)take(256);
  unsigned short* wg1t = (unsigned short*)take((size_t)F * H  * 2);
  unsigned short* wg2t = (unsigned short*)take((size_t)F * H  * 2);
  unsigned short* wl1t = (unsigned short*)take((size_t)H * H2 * 2);
  unsigned short* wr1t = (unsigned short*)take((size_t)H * H2 * 2);
  unsigned short* wl2t = (unsigned short*)take((size_t)H * H2 * 2);
  unsigned short* wr2t = (unsigned short*)take((size_t)H * H2 * 2);
  unsigned short* wm1t = (unsigned short*)take((size_t)H2 * H2 * 2);

  const dim3 blk(256);
  auto cdiv = [](size_t a, size_t b) { return (unsigned)((a + b - 1) / b); };
  auto cvt = [&](const float* s, size_t n) {
    f32_to_bf16<<<cdiv(n / 8, 256), blk, 0, stream>>>(s, abf, n / 8);
  };

  // ---- weight prep (bf16, transposed to [M][K]) ----
  transpose_to_bf16<<<cdiv((size_t)F * H, 256), blk, 0, stream>>>(Wg1, wg1t, F, H);
  transpose_to_bf16<<<cdiv((size_t)F * H, 256), blk, 0, stream>>>(Wg2, wg2t, F, H);
  transpose_to_bf16<<<cdiv((size_t)H * H2, 256), blk, 0, stream>>>(Wl1, wl1t, H, H2);
  transpose_to_bf16<<<cdiv((size_t)H * H2, 256), blk, 0, stream>>>(Wr1, wr1t, H, H2);
  transpose_to_bf16<<<cdiv((size_t)H * H2, 256), blk, 0, stream>>>(Wl2, wl2t, H, H2);
  transpose_to_bf16<<<cdiv((size_t)H * H2, 256), blk, 0, stream>>>(Wr2, wr2t, H, H2);
  transpose_to_bf16<<<cdiv((size_t)H2 * H2, 256), blk, 0, stream>>>(Wm1, wm1t, H2, H2);

  // ---- degrees / norms (deg starts at 1.0 = self-loop) ----
  fill_f32<<<cdiv(N, 256), blk, 0, stream>>>(degA, 1.0f, (size_t)N);
  fill_f32<<<cdiv(N, 256), blk, 0, stream>>>(degB, 1.0f, (size_t)N);
  fill_f32<<<1, blk, 0, stream>>>(sum, 0.0f, (size_t)1);
  deg_accum<<<cdiv(E, 256), blk, 0, stream>>>(dstA, degA, E);
  deg_accum<<<cdiv(E, 256), blk, 0, stream>>>(dstB, degB, E);
  deg_to_dis<<<cdiv(N, 256), blk, 0, stream>>>(degA, disA, N);
  deg_to_dis<<<cdiv(N, 256), blk, 0, stream>>>(degB, disB, N);

  const unsigned gR = cdiv(N, 128);   // row blocks (8 waves x 16 rows)
  // ---- GCN GEMMs: xw = x @ Wg ----
  cvt(x, (size_t)N * F);
  gemm_wmma_bf16<<<dim3(gR, H / 64), blk, 0, stream>>>(abf, wg1t, nullptr, buf1, N, F, H, 0, 0);
  cvt(x1, (size_t)N * F);
  gemm_wmma_bf16<<<dim3(gR, H / 64), blk, 0, stream>>>(abf, wg2t, nullptr, buf2, N, F, H, 0, 0);

  // ---- GCN aggregation: h=buf3, g=buf4 ----
  gcn_selfloop<<<N, blk, 0, stream>>>(buf1, disA, bg1, buf3);
  gcn_edge_scatter<<<E, blk, 0, stream>>>(buf1, disA, srcA, dstA, buf3);
  gcn_selfloop<<<N, blk, 0, stream>>>(buf2, disB, bg2, buf4);
  gcn_edge_scatter<<<E, blk, 0, stream>>>(buf2, disB, srcB, dstB, buf4);

  // ---- x0 = relu(h)+relu(g) -> buf1 ; g1 = relu(g)+g -> buf2 ----
  combine_streams<<<N, blk, 0, stream>>>(buf3, buf4, buf1, buf2);

  // ================= branch A (x0, graph A) =================
  fill_f32<<<cdiv((size_t)N * H, 256), blk, 0, stream>>>(buf3, 0.0f, (size_t)N * H);
  sage_edge_scatter<<<E, blk, 0, stream>>>(buf1, srcA, dstA, buf3);
  sage_mean<<<N, blk, 0, stream>>>(buf3, degA);
  cvt(buf3, (size_t)N * H);
  gemm_wmma_bf16<<<dim3(gR, H2 / 64), blk, 0, stream>>>(abf, wl1t, bl1,    buf4, N, H, H2, 0, 0);
  cvt(buf1, (size_t)N * H);
  gemm_wmma_bf16<<<dim3(gR, H2 / 64), blk, 0, stream>>>(abf, wr1t, nullptr, buf4, N, H, H2, 1, 1);
  cvt(buf4, (size_t)N * H2);
  gemm_wmma_bf16<<<dim3(gR, H2 / 64), blk, 0, stream>>>(abf, wm1t, bm1,    buf3, N, H2, H2, 0, 0);
  mlp_ln_dot<<<cdiv(N, 8), blk, 0, stream>>>(buf3, lnw, lnb, Wm2, bm2, sum, N);

  // ================= branch B (g1, graph B) =================
  fill_f32<<<cdiv((size_t)N * H, 256), blk, 0, stream>>>(buf3, 0.0f, (size_t)N * H);
  sage_edge_scatter<<<E, blk, 0, stream>>>(buf2, srcB, dstB, buf3);
  sage_mean<<<N, blk, 0, stream>>>(buf3, degB);
  cvt(buf3, (size_t)N * H);
  gemm_wmma_bf16<<<dim3(gR, H2 / 64), blk, 0, stream>>>(abf, wl2t, bl2,    buf4, N, H, H2, 0, 0);
  cvt(buf2, (size_t)N * H);
  gemm_wmma_bf16<<<dim3(gR, H2 / 64), blk, 0, stream>>>(abf, wr2t, nullptr, buf4, N, H, H2, 1, 1);
  cvt(buf4, (size_t)N * H2);
  gemm_wmma_bf16<<<dim3(gR, H2 / 64), blk, 0, stream>>>(abf, wm1t, bm1,    buf1, N, H2, H2, 0, 0);
  mlp_ln_dot<<<cdiv(N, 8), blk, 0, stream>>>(buf1, lnw, lnb, Wm2, bm2, sum, N);

  // ---- out = mean over 2N scalars ----
  finalize_mean<<<1, 1, 0, stream>>>(sum, out, 1.0f / (2.0f * (float)N));
  (void)n_in; (void)out_size; (void)ws_size;
}
